// MoELayer_44736379355137
// MI455X (gfx1250) — matrix operand order repaired and checked by reading, not compiled, not run on hardware
//
#include <hip/hip_runtime.h>
#include <cstdint>
#include <cstddef>

#define T_TOK 8192
#define DMODEL 1024
#define HID    2048
#define NEXP   8
#define KH     128   // hidden chunk width

typedef __attribute__((ext_vector_type(16))) __bf16 v16bf;
typedef __attribute__((ext_vector_type(8)))  float  v8f;

union Frag {
    uint4 u[2];
    v16bf v;
};

__device__ __forceinline__ unsigned short f2bf(float f) {
    unsigned int u = __float_as_uint(f);
    unsigned int r = (u + 0x7FFFu + ((u >> 16) & 1u)) >> 16;  // RNE
    return (unsigned short)r;
}

// gfx1250 async global->LDS copy (ASYNCcnt-tracked), per ISA 08_async_tensor §4
__device__ __forceinline__ void async_load_b128(unsigned lds_off,
                                                const void* gaddr) {
    asm volatile("global_load_async_to_lds_b128 %0, %1, off"
                 :: "v"(lds_off), "v"(gaddr)
                 : "memory");
}
__device__ __forceinline__ void wait_async0() {
    asm volatile("s_wait_asynccnt 0" ::: "memory");
}

// ---------------------------------------------------------------------------
// fp32 -> bf16 conversion (grid-stride, float4 granular)
// ---------------------------------------------------------------------------
__global__ void cvt_bf16_kernel(const float* __restrict__ in,
                                unsigned short* __restrict__ out, int n4) {
    const float4* in4 = (const float4*)in;
    ushort4* out4 = (ushort4*)out;
    int i = blockIdx.x * blockDim.x + threadIdx.x;
    int stride = gridDim.x * blockDim.x;
    for (; i < n4; i += stride) {
        float4 v = in4[i];
        ushort4 o;
        o.x = f2bf(v.x); o.y = f2bf(v.y); o.z = f2bf(v.z); o.w = f2bf(v.w);
        out4[i] = o;
    }
}

// ---------------------------------------------------------------------------
// Router: logits = x @ Wr, top-2 softmax -> dense combine[T, NEXP]
// ---------------------------------------------------------------------------
__global__ __launch_bounds__(256)
void router_kernel(const float* __restrict__ x, const float* __restrict__ Wr,
                   float* __restrict__ comb) {
    __shared__ float wr[DMODEL * NEXP];  // 32 KB
    const int tid = threadIdx.x;
    for (int i = tid; i < DMODEL * NEXP; i += 256) wr[i] = Wr[i];
    __syncthreads();

    const int t = blockIdx.x * 256 + tid;
    if (t >= T_TOK) return;

    const float4* xp = (const float4*)(x + (size_t)t * DMODEL);
    float l[NEXP];
#pragma unroll
    for (int e = 0; e < NEXP; ++e) l[e] = 0.0f;

    for (int i4 = 0; i4 < DMODEL / 4; ++i4) {
        float4 xv = xp[i4];
        const float* w0 = &wr[(i4 * 4) * NEXP];
#pragma unroll
        for (int e = 0; e < NEXP; ++e) {
            l[e] += xv.x * w0[e] + xv.y * w0[NEXP + e] +
                    xv.z * w0[2 * NEXP + e] + xv.w * w0[3 * NEXP + e];
        }
    }
    // top-2 (first occurrence on ties, matching lax.top_k)
    int i0 = 0; float v0 = l[0];
#pragma unroll
    for (int e = 1; e < NEXP; ++e) if (l[e] > v0) { v0 = l[e]; i0 = e; }
    int i1 = -1; float v1 = -3.4e38f;
#pragma unroll
    for (int e = 0; e < NEXP; ++e)
        if (e != i0 && l[e] > v1) { v1 = l[e]; i1 = e; }

    float e1 = expf(v1 - v0);
    float inv = 1.0f / (1.0f + e1);
    float w0s = inv, w1s = e1 * inv;

    float* c = comb + (size_t)t * NEXP;
#pragma unroll
    for (int e = 0; e < NEXP; ++e) c[e] = 0.0f;
    c[i0] = w0s;
    c[i1] = w1s;
}

// ---------------------------------------------------------------------------
// Fused MoE expert kernel. One block = 16 tokens, full 1024 output columns.
// 8 waves; per 128-wide hidden chunk, wave w computes BOTH the G and U
// 16x16 slices at columns kh+16w (A-fragment reused for 2 WMMAs);
// silu(g)*u*combine folded into H rows; each wave down-projects H into its
// private 128 output columns with WMMA bf16.
// ---------------------------------------------------------------------------
__global__ __launch_bounds__(256)
void moe_main_kernel(const unsigned short* __restrict__ xb,
                     const unsigned short* __restrict__ wgb,
                     const unsigned short* __restrict__ wub,
                     const unsigned short* __restrict__ wdb,
                     const float* __restrict__ comb,
                     float* __restrict__ out) {
    __shared__ __align__(16) unsigned short XA[16 * DMODEL];  // 32 KB bf16 X tile
    __shared__ __align__(16) float Gbuf[16 * KH];             // 8 KB
    __shared__ __align__(16) float Ubuf[16 * KH];             // 8 KB
    __shared__ __align__(16) unsigned short Hbuf[16 * KH];    // 4 KB bf16
    __shared__ float combS[16 * NEXP];
    __shared__ int eflag[NEXP];

    const int tid  = threadIdx.x;
    const int lane = tid & 31;
    const int w    = tid >> 5;        // wave id 0..7
    const int tb   = blockIdx.x * 16; // first token of tile

    // stage X tile (16 rows contiguous, 32 KB) via async global->LDS DMA
    {
        const char* src = (const char*)(xb + (size_t)tb * DMODEL);
        const unsigned base = (unsigned)(uintptr_t)(&XA[0]);
#pragma unroll
        for (int i = 0; i < 8; ++i) {
            int idx = tid + i * 256;  // 2048 x 16B = 32 KB
            async_load_b128(base + idx * 16, src + (size_t)idx * 16);
        }
    }
    if (tid < 16 * NEXP) combS[tid] = comb[(size_t)tb * NEXP + tid];
    wait_async0();
    __syncthreads();
    if (tid < NEXP) {
        int f = 0;
        for (int r = 0; r < 16; ++r) f |= (combS[r * NEXP + tid] != 0.0f);
        eflag[tid] = f;
    }
    __syncthreads();

    v8f outacc[8];
#pragma unroll
    for (int n = 0; n < 8; ++n)
#pragma unroll
        for (int v = 0; v < 8; ++v) outacc[n][v] = 0.0f;

    for (int e = 0; e < NEXP; ++e) {
        if (!eflag[e]) continue;  // block-uniform: safe with barriers
        const unsigned short* wge = wgb + (size_t)e * DMODEL * HID;
        const unsigned short* wue = wub + (size_t)e * DMODEL * HID;
        const unsigned short* wde = wdb + (size_t)e * HID * DMODEL;

        for (int kh = 0; kh < HID; kh += KH) {
            // ---- phase A: up-projection, wave's 16-wide G and U slices ----
            v8f gacc, uacc;
#pragma unroll
            for (int v = 0; v < 8; ++v) { gacc[v] = 0.0f; uacc[v] = 0.0f; }
            const int colb = kh + 16 * w;
            for (int kk = 0; kk < DMODEL; kk += 32) {
                Frag a, bg, bu;
                // A fragment (16x32 bf16): lanes 0-15 K[0..7,16..23],
                // lanes 16-31 K[8..15,24..31]
                const unsigned short* ap =
                    &XA[(lane & 15) * DMODEL + kk + (lane >> 4) * 8];
                a.u[0] = *(const uint4*)ap;
                a.u[1] = *(const uint4*)(ap + 16);
                // B fragments (32x16 bf16): lane = N col, 16 K rows in table order
                const size_t boff = (size_t)kk * HID + colb + (lane & 15);
                const unsigned short* bgp = wge + boff;
                const unsigned short* bup = wue + boff;
#pragma unroll
                for (int s = 0; s < 16; ++s) {
                    int k = (s < 8) ? (8 * (lane >> 4) + s) : (8 + 8 * (lane >> 4) + s);
                    ((unsigned short*)&bg)[s] = bgp[(size_t)k * HID];
                    ((unsigned short*)&bu)[s] = bup[(size_t)k * HID];
                }
                __builtin_prefetch(bgp + 32 * HID, 0, 1);  // global_prefetch_b8
                gacc = __builtin_amdgcn_wmma_f32_16x16x32_bf16(
                    false, a.v, false, bg.v, (short)0, gacc, false, false);
                uacc = __builtin_amdgcn_wmma_f32_16x16x32_bf16(
                    false, a.v, false, bu.v, (short)0, uacc, false, false);
            }
#pragma unroll
            for (int v = 0; v < 8; ++v) {
                int row = v + ((lane >> 4) << 3);
                int c   = 16 * w + (lane & 15);
                Gbuf[row * KH + c] = gacc[v];
                Ubuf[row * KH + c] = uacc[v];
            }
            __syncthreads();

            // ---- phase B: h = silu(g)*u * combine[row], to bf16 LDS -------
#pragma unroll
            for (int jj = 0; jj < 8; ++jj) {
                int idx = tid + jj * 256;          // 16*128 = 2048 elements
                int row = idx >> 7;                // /KH
                float g = Gbuf[idx];
                float u = Ubuf[idx];
                float s = g / (1.0f + __expf(-g));
                float h = s * u * combS[row * NEXP + e];
                Hbuf[idx] = f2bf(h);
            }
            __syncthreads();

            // ---- phase C: outacc += H(16x128) @ Wd[kh:kh+128, wave cols] --
#pragma unroll
            for (int ks = 0; ks < KH; ks += 32) {
                Frag ha;
                const unsigned short* hp =
                    &Hbuf[(lane & 15) * KH + ks + (lane >> 4) * 8];
                ha.u[0] = *(const uint4*)hp;
                ha.u[1] = *(const uint4*)(hp + 16);
                const unsigned short* dbase =
                    wde + (size_t)(kh + ks) * DMODEL + w * 128 + (lane & 15);
#pragma unroll
                for (int n = 0; n < 8; ++n) {
                    Frag db;
                    const unsigned short* dp = dbase + 16 * n;
#pragma unroll
                    for (int s = 0; s < 16; ++s) {
                        int k = (s < 8) ? (8 * (lane >> 4) + s) : (8 + 8 * (lane >> 4) + s);
                        ((unsigned short*)&db)[s] = dp[(size_t)k * DMODEL];
                    }
                    outacc[n] = __builtin_amdgcn_wmma_f32_16x16x32_bf16(
                        false, ha.v, false, db.v, (short)0, outacc[n],
                        false, false);
                }
            }
            __syncthreads();
        }
    }

    // ---- write final output (fp32), full coverage of 16x1024 tile ---------
#pragma unroll
    for (int n = 0; n < 8; ++n) {
#pragma unroll
        for (int v = 0; v < 8; ++v) {
            int row = v + ((lane >> 4) << 3);
            int col = w * 128 + 16 * n + (lane & 15);
            out[(size_t)(tb + row) * DMODEL + col] = outacc[n][v];
        }
    }
}

// ---------------------------------------------------------------------------
extern "C" void kernel_launch(void* const* d_in, const int* in_sizes, int n_in,
                              void* d_out, int out_size, void* d_ws,
                              size_t ws_size, hipStream_t stream) {
    (void)in_sizes; (void)n_in; (void)out_size; (void)ws_size;
    const float* x  = (const float*)d_in[0];
    const float* Wr = (const float*)d_in[1];
    const float* Wg = (const float*)d_in[2];
    const float* Wu = (const float*)d_in[3];
    const float* Wd = (const float*)d_in[4];
    float* out = (float*)d_out;

    // workspace layout (bf16 mirrors + combine table), ~118 MB total
    unsigned short* xb  = (unsigned short*)d_ws;
    unsigned short* wgb = xb  + (size_t)T_TOK * DMODEL;
    unsigned short* wub = wgb + (size_t)NEXP * DMODEL * HID;
    unsigned short* wdb = wub + (size_t)NEXP * DMODEL * HID;
    float* comb = (float*)(wdb + (size_t)NEXP * HID * DMODEL);

    dim3 blk(256);
    cvt_bf16_kernel<<<2048, blk, 0, stream>>>(x,  xb,  T_TOK * DMODEL / 4);
    cvt_bf16_kernel<<<4096, blk, 0, stream>>>(Wg, wgb, NEXP * DMODEL * HID / 4);
    cvt_bf16_kernel<<<4096, blk, 0, stream>>>(Wu, wub, NEXP * DMODEL * HID / 4);
    cvt_bf16_kernel<<<4096, blk, 0, stream>>>(Wd, wdb, NEXP * HID * DMODEL / 4);
    router_kernel<<<T_TOK / 256, blk, 0, stream>>>(x, Wr, comb);
    moe_main_kernel<<<T_TOK / 16, blk, 0, stream>>>(xb, wgb, wub, wdb, comb, out);
}
